// attention_decoding_71803263254779
// MI455X (gfx1250) — matrix operand order repaired and checked by reading, not compiled, not run on hardware
//
#include <hip/hip_runtime.h>
#include <hip/hip_bf16.h>

// ---------------- constants (from reference) ----------------
#define FSZ   128
#define CNUMC 2048
#define ANUMC 16
#define BATCHC 64
#define NTOT  2080            // CNUM + 2*ANUM
#define A2F   256
#define A3F   384
#define NAG   17               // 16 agents + free agent
#define SCALE_F 0.08838834764831845f   // 1/sqrt(128)

// ---------------- WMMA types ----------------
typedef __attribute__((ext_vector_type(16))) __bf16 v16bf;
typedef __attribute__((ext_vector_type(8)))  float  v8f;
typedef __attribute__((ext_vector_type(4)))  unsigned int u32x4;

union Frag16 { v16bf v; u32x4 q[2]; };

static __device__ inline v8f wmma_bf16(const Frag16& a, const Frag16& b, v8f c) {
  return __builtin_amdgcn_wmma_f32_16x16x32_bf16(false, a.v, false, b.v,
                                                 (short)0, c, false, false);
}

static __device__ inline unsigned short f2bf(float f) {
  unsigned int u = __float_as_uint(f);
  unsigned int r = u + 0x7FFFu + ((u >> 16) & 1u);   // round-to-nearest-even
  return (unsigned short)(r >> 16);
}

// ---------------- Kernel A: per (b,feature) reductions over N ----------------
__global__ void kA_reduce(const float* __restrict__ nf,
                          float* __restrict__ glb, float* __restrict__ ave) {
  int bk = blockIdx.x;                 // BATCH*FSZ blocks
  int b = bk >> 7, k = bk & 127;
  const float* row = nf + ((size_t)b * FSZ + k) * NTOT;
  int tid = threadIdx.x;               // 128 threads
  float mx = -INFINITY, sm = 0.f;
  for (int c = tid; c < CNUMC; c += 128) { float v = row[c]; mx = fmaxf(mx, v); sm += v; }
  __shared__ float smx[128], ssm[128];
  smx[tid] = mx; ssm[tid] = sm; __syncthreads();
  for (int s = 64; s > 0; s >>= 1) {
    if (tid < s) { smx[tid] = fmaxf(smx[tid], smx[tid + s]); ssm[tid] += ssm[tid + s]; }
    __syncthreads();
  }
  if (tid == 0) {
    float cmax = smx[0], csum = ssm[0];
    float smax = -INFINITY, ssum = 0.f, emax = -INFINITY, esum = 0.f;
    for (int i = 0; i < ANUMC; i++) {
      float v = row[CNUMC + i];          smax = fmaxf(smax, v); ssum += v;
      float w = row[CNUMC + ANUMC + i];  emax = fmaxf(emax, w); esum += w;
    }
    float* g = glb + (size_t)b * 512;
    float* a = ave + (size_t)b * 512;
    g[k]       = fmaxf(cmax, fmaxf(smax, emax));
    g[128 + k] = cmax; g[256 + k] = smax; g[384 + k] = emax;
    a[k]       = (csum + ssum + esum) * (1.f / (float)NTOT);
    a[128 + k] = csum * (1.f / (float)CNUMC);
    a[256 + k] = ssum * (1.f / (float)ANUMC);
    a[384 + k] = esum * (1.f / (float)ANUMC);
  }
}

// ---------------- Kernel W: weight conversion to bf16 ----------------
__global__ void kW_convert(const float* __restrict__ Wc, const float* __restrict__ Wp,
                           unsigned short* __restrict__ Wc2bf,
                           unsigned short* __restrict__ Wpbf) {
  int idx = blockIdx.x * blockDim.x + threadIdx.x;
  if (idx < A3F * FSZ) {                 // Wc2bf[o][k] = Wc[o][128+k]
    int o = idx >> 7, k = idx & 127;
    Wc2bf[idx] = f2bf(Wc[(size_t)o * A2F + FSZ + k]);
  }
  int j = idx - A3F * FSZ;
  if (j >= 0 && j < FSZ * FSZ) Wpbf[j] = f2bf(Wp[j]);
}

// ---------------- Kernel B: per-batch small dots ----------------
__global__ void kB_small(const float* __restrict__ nf,
                         const float* __restrict__ glb, const float* __restrict__ ave,
                         const float* __restrict__ Wg, const float* __restrict__ bg,
                         const float* __restrict__ Wc, const float* __restrict__ bc,
                         const float* __restrict__ Wa, const float* __restrict__ ba,
                         float* __restrict__ cbias, unsigned short* __restrict__ af_bf) {
  int b = blockIdx.x, tid = threadIdx.x;  // 256 threads
  __shared__ float deglb[128], freeag[128];
  __shared__ float ast[32 * 128];         // rows 0..15 a_start, 16..31 a_end
  const float* g = glb + (size_t)b * 512;
  const float* a = ave + (size_t)b * 512;
  if (tid < 128) {
    float s = bg[tid];
    const float* wr = Wg + (size_t)tid * 512;
    for (int j = 0; j < 512; j++) s += g[j] * wr[j];
    deglb[tid] = s;
  } else {
    int o = tid - 128;
    float s = bg[o];
    const float* wr = Wg + (size_t)o * 512;
    for (int j = 0; j < 512; j++) s += a[j] * wr[j];
    freeag[o] = s;
  }
  for (int idx = tid; idx < 128 * 32; idx += 256) {
    int k = idx >> 5, aa = idx & 31;
    ast[aa * 128 + k] = nf[((size_t)b * FSZ + k) * NTOT + CNUMC + aa];
  }
  __syncthreads();
  for (int oo = tid; oo < A3F; oo += 256) {
    float s = bc[oo];
    const float* wr = Wc + (size_t)oo * A2F;
    for (int k = 0; k < 128; k++) s += wr[k] * deglb[k];
    cbias[(size_t)b * A3F + oo] = s;
  }
  for (int idx = tid; idx < 32 * 128; idx += 256) {
    int aa = idx >> 7, o = idx & 127;
    float v = 0.f;
    if (aa < 16) {
      const float* wr = Wa + (size_t)o * A3F;
      float s = ba[o];
      const float* as0 = &ast[aa * 128];
      const float* ae0 = &ast[(16 + aa) * 128];
      for (int k = 0; k < 128; k++) s += wr[k] * deglb[k];
      for (int k = 0; k < 128; k++) s += wr[128 + k] * as0[k];
      for (int k = 0; k < 128; k++) s += wr[256 + k] * ae0[k];
      v = s;
    } else if (aa == 16) {
      v = freeag[o];
    }
    af_bf[((size_t)b * 32 + aa) * 128 + o] = f2bf(v);
  }
}

// ---------------- Kernel C: big GEMM  xyz = city @ Wc2^T + cbias ----------------
#define LDA_C 136
__global__ void kC_xyz(const float* __restrict__ nf,
                       const unsigned short* __restrict__ Wc2bf,
                       const float* __restrict__ cbias,
                       unsigned short* __restrict__ X,
                       unsigned short* __restrict__ YT,
                       unsigned short* __restrict__ Z) {
  int cb = blockIdx.x * 128;             // city tile
  int ob = blockIdx.y * 64;              // output-feature tile within 384
  int b  = blockIdx.z;
  int tid = threadIdx.x;
  __shared__ unsigned short cityT[128 * LDA_C];   // [c][k] bf16, k-contiguous
  const float* src = nf + (size_t)b * FSZ * NTOT + cb;
  for (int idx = tid; idx < 128 * 128; idx += 256) {
    int k = idx >> 7, c = idx & 127;
    cityT[c * LDA_C + k] = f2bf(src[(size_t)k * NTOT + c]);
  }
  __syncthreads();
  int wid = tid >> 5, lane = tid & 31, half = lane >> 4, ln = lane & 15;
  int wm = (wid & 3) * 32;               // 4 waves along M (128 cities)
  int wn = (wid >> 2) * 32;              // 2 waves along N (64 feats)
  v8f acc[2][2] = {};
  for (int kk = 0; kk < 128; kk += 32) {
    Frag16 a0, a1, b0, b1;
    int ab0 = (wm + ln) * LDA_C + kk + 8 * half;
    a0.q[0] = *(const u32x4*)&cityT[ab0];  a0.q[1] = *(const u32x4*)&cityT[ab0 + 16];
    int ab1 = (wm + 16 + ln) * LDA_C + kk + 8 * half;
    a1.q[0] = *(const u32x4*)&cityT[ab1];  a1.q[1] = *(const u32x4*)&cityT[ab1 + 16];
    const unsigned short* w0 = Wc2bf + (size_t)(ob + wn + ln) * FSZ + kk + 8 * half;
    b0.q[0] = ((const u32x4*)w0)[0];       b0.q[1] = ((const u32x4*)w0)[2];
    const unsigned short* w1 = Wc2bf + (size_t)(ob + wn + 16 + ln) * FSZ + kk + 8 * half;
    b1.q[0] = ((const u32x4*)w1)[0];       b1.q[1] = ((const u32x4*)w1)[2];
    acc[0][0] = wmma_bf16(a0, b0, acc[0][0]);
    acc[0][1] = wmma_bf16(a0, b1, acc[0][1]);
    acc[1][0] = wmma_bf16(a1, b0, acc[1][0]);
    acc[1][1] = wmma_bf16(a1, b1, acc[1][1]);
  }
  const float* cbb = cbias + (size_t)b * A3F;
  for (int j = 0; j < 2; j++) {
    int o3 = ob + wn + j * 16 + ln;            // 0..383
    float bias = cbb[o3];
    for (int i = 0; i < 2; i++) {
      for (int r = 0; r < 8; r++) {
        int c = cb + wm + i * 16 + r + 8 * half;
        unsigned short h = f2bf(acc[i][j][r] + bias);
        if (o3 < 128)       X[(((size_t)b * CNUMC + c) << 7) + o3] = h;
        else if (o3 < 256)  YT[((size_t)b * FSZ + (o3 - 128)) * CNUMC + c] = h;
        else                Z[(((size_t)b * CNUMC + c) << 7) + (o3 - 256)] = h;
      }
    }
  }
}

// ---------------- Kernel D: flash attention + Wp projection ----------------
#define LDS_S 132
#define LDA_D 136
__global__ void kD_attn(const unsigned short* __restrict__ af_bf,
                        const unsigned short* __restrict__ X,
                        const unsigned short* __restrict__ YT,
                        const unsigned short* __restrict__ Wpbf,
                        const float* __restrict__ bp,
                        unsigned short* __restrict__ naf_bf) {
  int b = blockIdx.x, tid = threadIdx.x;  // 256 threads / 8 waves
  __shared__ unsigned short afs[32 * LDA_D];   // af (later reused as normalized new_af)
  __shared__ unsigned short Ps[32 * LDA_D];    // exp(ucj - m) bf16
  __shared__ float S[32 * LDS_S];              // ucj chunk fp32
  __shared__ float m_s[32], l_s[32], rs_s[32];
  for (int idx = tid; idx < 32 * 128; idx += 256) {
    int aa = idx >> 7, k = idx & 127;
    afs[aa * LDA_D + k] = af_bf[((size_t)b * 32 + aa) * 128 + k];
  }
  if (tid < 32) { m_s[tid] = -INFINITY; l_s[tid] = 0.f; }
  __syncthreads();
  int wid = tid >> 5, lane = tid & 31, half = lane >> 4, ln = lane & 15;
  int fw = wid * 16;                     // this wave's 16-wide column slice
  v8f acc0 = {}, acc1 = {};
  for (int cc = 0; cc < CNUMC; cc += 128) {
    // 1) ucj tile: 32 agents x 16 cities per wave
    v8f u0 = {}, u1 = {};
    for (int kk = 0; kk < 128; kk += 32) {
      Frag16 a0, a1, bx;
      int ab0 = ln * LDA_D + kk + 8 * half;
      a0.q[0] = *(const u32x4*)&afs[ab0];  a0.q[1] = *(const u32x4*)&afs[ab0 + 16];
      int ab1 = (16 + ln) * LDA_D + kk + 8 * half;
      a1.q[0] = *(const u32x4*)&afs[ab1];  a1.q[1] = *(const u32x4*)&afs[ab1 + 16];
      const unsigned short* xr = X + (((size_t)b * CNUMC + cc + fw + ln) << 7) + kk + 8 * half;
      bx.q[0] = ((const u32x4*)xr)[0];     bx.q[1] = ((const u32x4*)xr)[2];
      u0 = wmma_bf16(a0, bx, u0);
      u1 = wmma_bf16(a1, bx, u1);
    }
    for (int r = 0; r < 8; r++) {
      S[(r + 8 * half) * LDS_S + fw + ln]      = u0[r] * SCALE_F;
      S[(16 + r + 8 * half) * LDS_S + fw + ln] = u1[r] * SCALE_F;
    }
    __syncthreads();
    // 2) flash softmax update (one thread per agent row)
    if (tid < 32) {
      float mo = m_s[tid], mx = mo;
      const float* Sr = &S[tid * LDS_S];
      for (int c = 0; c < 128; c++) mx = fmaxf(mx, Sr[c]);
      float rs = __expf(mo - mx);          // -inf -> 0 on first chunk
      float sum = 0.f;
      unsigned short* Pr = &Ps[tid * LDA_D];
      for (int c = 0; c < 128; c++) { float p = __expf(Sr[c] - mx); sum += p; Pr[c] = f2bf(p); }
      m_s[tid] = mx; l_s[tid] = l_s[tid] * rs + sum; rs_s[tid] = rs;
    }
    __syncthreads();
    // 3) rescale accumulator, then acc += P @ Ychunk
    for (int r = 0; r < 8; r++) {
      acc0[r] *= rs_s[r + 8 * half];
      acc1[r] *= rs_s[16 + r + 8 * half];
    }
    for (int kk = 0; kk < 128; kk += 32) {
      Frag16 p0, p1, by;
      int pb0 = ln * LDA_D + kk + 8 * half;
      p0.q[0] = *(const u32x4*)&Ps[pb0];  p0.q[1] = *(const u32x4*)&Ps[pb0 + 16];
      int pb1 = (16 + ln) * LDA_D + kk + 8 * half;
      p1.q[0] = *(const u32x4*)&Ps[pb1];  p1.q[1] = *(const u32x4*)&Ps[pb1 + 16];
      const unsigned short* yr = YT + ((size_t)b * FSZ + fw + ln) * CNUMC + cc + kk + 8 * half;
      by.q[0] = ((const u32x4*)yr)[0];    by.q[1] = ((const u32x4*)yr)[2];
      acc0 = wmma_bf16(p0, by, acc0);
      acc1 = wmma_bf16(p1, by, acc1);
    }
    __syncthreads();
  }
  // normalize -> reuse afs as bf16 new_af buffer
  for (int r = 0; r < 8; r++) {
    int r0 = r + 8 * half, r1 = 16 + r + 8 * half;
    afs[r0 * LDA_D + fw + ln] = f2bf(acc0[r] / l_s[r0]);
    afs[r1 * LDA_D + fw + ln] = f2bf(acc1[r] / l_s[r1]);
  }
  __syncthreads();
  // projection: naf = new_af @ Wp^T + bp
  v8f n0 = {}, n1 = {};
  for (int kk = 0; kk < 128; kk += 32) {
    Frag16 a0, a1, bw;
    int ab0 = ln * LDA_D + kk + 8 * half;
    a0.q[0] = *(const u32x4*)&afs[ab0];  a0.q[1] = *(const u32x4*)&afs[ab0 + 16];
    int ab1 = (16 + ln) * LDA_D + kk + 8 * half;
    a1.q[0] = *(const u32x4*)&afs[ab1];  a1.q[1] = *(const u32x4*)&afs[ab1 + 16];
    const unsigned short* wr = Wpbf + (size_t)(fw + ln) * FSZ + kk + 8 * half;
    bw.q[0] = ((const u32x4*)wr)[0];     bw.q[1] = ((const u32x4*)wr)[2];
    n0 = wmma_bf16(a0, bw, n0);
    n1 = wmma_bf16(a1, bw, n1);
  }
  float bpo = bp[fw + ln];
  for (int r = 0; r < 8; r++) {
    naf_bf[((size_t)b * 32 + (r + 8 * half)) * 128 + fw + ln]      = f2bf(n0[r] + bpo);
    naf_bf[((size_t)b * 32 + (16 + r + 8 * half)) * 128 + fw + ln] = f2bf(n1[r] + bpo);
  }
}

// ---------------- Kernel E: logits + softmax over agents ----------------
__global__ void kE_logits(const unsigned short* __restrict__ naf_bf,
                          const unsigned short* __restrict__ Z,
                          float* __restrict__ out) {
  int cb = blockIdx.x * 128;
  int b  = blockIdx.y;
  int tid = threadIdx.x;
  __shared__ float Lg[128 * 33];
  int wid = tid >> 5, lane = tid & 31, half = lane >> 4, ln = lane & 15;
  int c = cb + wid * 16 + ln;
  v8f t0 = {}, t1 = {};
  for (int kk = 0; kk < 128; kk += 32) {
    Frag16 a0, a1, bz;
    const unsigned short* n0p = naf_bf + (size_t)b * 4096 + ln * 128 + kk + 8 * half;
    a0.q[0] = ((const u32x4*)n0p)[0];        a0.q[1] = *(const u32x4*)(n0p + 16);
    const unsigned short* n1p = naf_bf + (size_t)b * 4096 + (16 + ln) * 128 + kk + 8 * half;
    a1.q[0] = ((const u32x4*)n1p)[0];        a1.q[1] = *(const u32x4*)(n1p + 16);
    const unsigned short* zr = Z + (((size_t)b * CNUMC + c) << 7) + kk + 8 * half;
    bz.q[0] = ((const u32x4*)zr)[0];         bz.q[1] = ((const u32x4*)zr)[2];
    t0 = wmma_bf16(a0, bz, t0);
    t1 = wmma_bf16(a1, bz, t1);
  }
  int clocal = wid * 16 + ln;
  for (int r = 0; r < 8; r++) {
    Lg[clocal * 33 + (r + 8 * half)]      = tanhf(t0[r] * SCALE_F) * 10.f;
    Lg[clocal * 33 + (16 + r + 8 * half)] = tanhf(t1[r] * SCALE_F) * 10.f;
  }
  __syncthreads();
  if (tid < 128) {
    const float* Lr = &Lg[tid * 33];
    float mx = -INFINITY;
    for (int a2 = 0; a2 < NAG; a2++) mx = fmaxf(mx, Lr[a2]);
    float e[NAG], sum = 0.f;
    for (int a2 = 0; a2 < NAG; a2++) { e[a2] = __expf(Lr[a2] - mx); sum += e[a2]; }
    float inv = 1.f / sum;
    float* op = out + ((size_t)b * CNUMC + cb + tid) * NAG;
    for (int a2 = 0; a2 < NAG; a2++) op[a2] = e[a2] * inv;
  }
}

// ---------------- launch ----------------
extern "C" void kernel_launch(void* const* d_in, const int* in_sizes, int n_in,
                              void* d_out, int out_size, void* d_ws, size_t ws_size,
                              hipStream_t stream) {
  (void)in_sizes; (void)n_in; (void)out_size; (void)ws_size;
  const float* nf = (const float*)d_in[0];
  const float* Wg = (const float*)d_in[1];
  const float* bg = (const float*)d_in[2];
  const float* Wc = (const float*)d_in[3];
  const float* bc = (const float*)d_in[4];
  const float* Wa = (const float*)d_in[5];
  const float* ba = (const float*)d_in[6];
  const float* Wp = (const float*)d_in[7];
  const float* bp = (const float*)d_in[8];
  float* out = (float*)d_out;

  char* ws = (char*)d_ws;
  size_t off = 0;
  auto alloc = [&](size_t bytes) { size_t o = off; off = (off + bytes + 255) & ~(size_t)255; return o; };
  float*          glb    = (float*)(ws + alloc((size_t)BATCHC * 512 * 4));
  float*          ave    = (float*)(ws + alloc((size_t)BATCHC * 512 * 4));
  float*          cbias  = (float*)(ws + alloc((size_t)BATCHC * A3F * 4));
  unsigned short* af_bf  = (unsigned short*)(ws + alloc((size_t)BATCHC * 32 * 128 * 2));
  unsigned short* naf_bf = (unsigned short*)(ws + alloc((size_t)BATCHC * 32 * 128 * 2));
  unsigned short* Wc2bf  = (unsigned short*)(ws + alloc((size_t)A3F * FSZ * 2));
  unsigned short* Wpbf   = (unsigned short*)(ws + alloc((size_t)FSZ * FSZ * 2));
  unsigned short* X      = (unsigned short*)(ws + alloc((size_t)BATCHC * CNUMC * FSZ * 2));
  unsigned short* YT     = (unsigned short*)(ws + alloc((size_t)BATCHC * CNUMC * FSZ * 2));
  unsigned short* Z      = (unsigned short*)(ws + alloc((size_t)BATCHC * CNUMC * FSZ * 2));

  kA_reduce<<<dim3(BATCHC * FSZ), dim3(128), 0, stream>>>(nf, glb, ave);
  kW_convert<<<dim3((A3F * FSZ + FSZ * FSZ + 255) / 256), dim3(256), 0, stream>>>(Wc, Wp, Wc2bf, Wpbf);
  kB_small<<<dim3(BATCHC), dim3(256), 0, stream>>>(nf, glb, ave, Wg, bg, Wc, bc, Wa, ba, cbias, af_bf);
  kC_xyz<<<dim3(CNUMC / 128, A3F / 64, BATCHC), dim3(256), 0, stream>>>(nf, Wc2bf, cbias, X, YT, Z);
  kD_attn<<<dim3(BATCHC), dim3(256), 0, stream>>>(af_bf, X, YT, Wpbf, bp, naf_bf);
  kE_logits<<<dim3(CNUMC / 128, BATCHC), dim3(256), 0, stream>>>(naf_bf, Z, out);
}
